// Head_14010183319898
// MI455X (gfx1250) — matrix operand order repaired and checked by reading, not compiled, not run on hardware
//
#include <hip/hip_runtime.h>
#include <hip/hip_bf16.h>

typedef __attribute__((ext_vector_type(4)))  float  v4f;
typedef __attribute__((ext_vector_type(8)))  float  v8f;
typedef __attribute__((ext_vector_type(16))) __bf16 v16bf;
typedef __attribute__((ext_vector_type(8)))  __bf16 v8bf;

#define WMMA_BF16(A, B, C) \
    __builtin_amdgcn_wmma_f32_16x16x32_bf16(false, (A), false, (B), (short)0, (C), false, false)

union V16 { v16bf v; v8bf h[2]; unsigned short s[16]; };

// round-to-nearest-even f32 -> bf16 (inputs are finite here; skip NaN path)
__device__ __forceinline__ unsigned short f2bf(float f) {
    union { float f; unsigned u; } x; x.f = f;
    return (unsigned short)((x.u + 0x7fffu + ((x.u >> 16) & 1u)) >> 16);
}

// ---------------------------------------------------------------------------
// Kernel 0: W[1024][64] f32 -> Wt[3][64][1024] bf16 (transposed, bf16)
// ---------------------------------------------------------------------------
__global__ void wt_kernel(const float* __restrict__ Wq, const float* __restrict__ Wk,
                          const float* __restrict__ Wv, __bf16* __restrict__ Wt) {
    int idx = blockIdx.x * 256 + threadIdx.x;           // 3*64*1024 = 196608
    if (idx >= 3 * 64 * 1024) return;
    int m   = idx >> 16;
    int rem = idx & 65535;
    int d   = rem >> 10;
    int k   = rem & 1023;
    const float* W = (m == 0) ? Wq : ((m == 1) ? Wk : Wv);
    ((unsigned short*)Wt)[idx] = f2bf(W[k * 64 + d]);
}

// ---------------------------------------------------------------------------
// Kernel 1: projections (bf16 WMMA, f32 accum) + RoPE.
// One wave handles 16 rows of x. Q,K stored row-major bf16; V stored as
// Vt[b][64][2048] bf16 (transposed) for contiguous attention B-operands.
// x is streamed with non-temporal loads (read-once; keep L2 for Q/K/Vt).
// ---------------------------------------------------------------------------
__global__ __launch_bounds__(128) void proj_rope_kernel(
    const float* __restrict__ x, const float* __restrict__ fcos,
    const float* __restrict__ fsin, const __bf16* __restrict__ Wt,
    __bf16* __restrict__ Qb, __bf16* __restrict__ Kb, __bf16* __restrict__ Vt) {

    const int lane = threadIdx.x & 31;
    const int wv   = threadIdx.x >> 5;
    const int l15  = lane & 15;
    const int hi   = (lane >> 4) & 1;
    const int tile = blockIdx.x * 4 + wv;               // 0..1023
    const int b    = tile >> 7;                          // / (2048/16)
    const int trow = (tile & 127) << 4;

    const float* xrow = x + ((size_t)b * 2048 + trow + l15) * 1024;
    const __bf16* WtQ = Wt;
    const __bf16* WtK = Wt + 64 * 1024;
    const __bf16* WtV = Wt + 2 * 64 * 1024;

    v8f aq[4], ak[4], av[4];
#pragma unroll
    for (int i = 0; i < 4; i++) { aq[i] = (v8f){}; ak[i] = (v8f){}; av[i] = (v8f){}; }

    for (int k0 = 0; k0 < 1024; k0 += 32) {
        // A operand: x tile 16x32 (K = e + 16*hi, matching B)
        const float* p0 = xrow + k0 + 16 * hi;
        const float* p1 = xrow + k0 + 16 * hi + 8;
        v4f xa = __builtin_nontemporal_load((const v4f*)p0);
        v4f xb = __builtin_nontemporal_load((const v4f*)(p0 + 4));
        v4f xc = __builtin_nontemporal_load((const v4f*)p1);
        v4f xd = __builtin_nontemporal_load((const v4f*)(p1 + 4));
        V16 A;
#pragma unroll
        for (int e = 0; e < 4; e++) {
            A.s[e]      = f2bf(xa[e]);
            A.s[4 + e]  = f2bf(xb[e]);
            A.s[8 + e]  = f2bf(xc[e]);
            A.s[12 + e] = f2bf(xd[e]);
        }

#pragma unroll
        for (int dn = 0; dn < 4; dn++) {
            // B operand (same k order as A: K = e + 16*hi, col = lane&15): contiguous in Wt
            size_t wo = (size_t)(dn * 16 + l15) * 1024 + k0 + 16 * hi;
            v16bf bq = *(const v16bf*)(WtQ + wo);
            v16bf bk = *(const v16bf*)(WtK + wo);
            v16bf bv = *(const v16bf*)(WtV + wo);
            aq[dn] = WMMA_BF16(A.v, bq, aq[dn]);
            ak[dn] = WMMA_BF16(A.v, bk, ak[dn]);
            av[dn] = WMMA_BF16(A.v, bv, av[dn]);
        }
    }

    // RoPE in C-layout (row = r+8*hi, col d = dn*16 + l15); pair partner is lane^1
#pragma unroll
    for (int dn = 0; dn < 4; dn++) {
        const int d = dn * 16 + l15;
        const int j = d >> 1;
#pragma unroll
        for (int r = 0; r < 8; r++) {
            const int tg = trow + r + 8 * hi;
            float c = fcos[tg * 32 + j];
            float s = fsin[tg * 32 + j];
            float vq = aq[dn][r]; float pq = __shfl_xor(vq, 1);
            float vk = ak[dn][r]; float pk = __shfl_xor(vk, 1);
            float rq = (d & 1) ? (pq * s + vq * c) : (vq * c - pq * s);
            float rk = (d & 1) ? (pk * s + vk * c) : (vk * c - pk * s);
            size_t o = ((size_t)b * 2048 + tg) * 64 + d;
            ((unsigned short*)Qb)[o] = f2bf(rq);
            ((unsigned short*)Kb)[o] = f2bf(rk);
        }
        // V transposed; pack bf16 pairs (t contiguous over r) into b32 stores
        unsigned short vs[8];
#pragma unroll
        for (int r = 0; r < 8; r++) vs[r] = f2bf(av[dn][r]);
        size_t vo = ((size_t)b * 64 + d) * 2048 + trow + 8 * hi;
#pragma unroll
        for (int r = 0; r < 8; r += 2) {
            unsigned pack = (unsigned)vs[r] | ((unsigned)vs[r + 1] << 16);
            *(unsigned*)((unsigned short*)Vt + vo + r) = pack;
        }
    }
}

// ---------------------------------------------------------------------------
// Kernel 2: wave-level flash attention. 16 queries/wave, 32-key tiles,
// exp2-domain online softmax; P converted C-layout -> A-layout through LDS;
// next K/V tile prefetched (global_prefetch_b8) to overlap with softmax chain.
// ---------------------------------------------------------------------------
__global__ __launch_bounds__(128) void flash_attn_kernel(
    const __bf16* __restrict__ Qb, const __bf16* __restrict__ Kb,
    const __bf16* __restrict__ Vt, float* __restrict__ out) {

    __shared__ __align__(32) unsigned short sP[4][16 * 32];   // per-wave P tile (bf16)

    const int lane  = threadIdx.x & 31;
    const int wv    = threadIdx.x >> 5;
    const int l15   = lane & 15;
    const int hi    = (lane >> 4) & 1;
    const int tile  = blockIdx.x * 4 + wv;                    // 0..1023
    const int b     = tile >> 7;
    const int qbase = (tile & 127) << 4;

    const __bf16* Qbb = Qb + (size_t)b * 2048 * 64;
    const __bf16* Kbb = Kb + (size_t)b * 2048 * 64;
    const __bf16* Vtb = Vt + (size_t)b * 64 * 2048;

    // Preload both Q A-operands (full 64-wide row per lane), reused all key tiles
    const __bf16* Qrow = Qbb + (size_t)(qbase + l15) * 64;
    V16 aq0, aq1;
    aq0.h[0] = *(const v8bf*)(Qrow + 16 * hi);
    aq0.h[1] = *(const v8bf*)(Qrow + 8 + 16 * hi);
    aq1.h[0] = *(const v8bf*)(Qrow + 32 + 16 * hi);
    aq1.h[1] = *(const v8bf*)(Qrow + 40 + 16 * hi);

    v8f O[4];
#pragma unroll
    for (int i = 0; i < 4; i++) O[i] = (v8f){};
    float mrow[8], lrow[8];
#pragma unroll
    for (int r = 0; r < 8; r++) { mrow[r] = -__builtin_inff(); lrow[r] = 0.f; }

    unsigned short* myP = &sP[wv][0];
    // exp2-domain logits: fold 1024^-0.5 * log2(e) into one constant
    const float SCL2 = 0.03125f * 1.4426950408889634f;
    const int nkt = (qbase + 16 + 31) >> 5;         // key tiles of 32 (causal bound)

    for (int kt = 0; kt < nkt; kt++) {
        const int kb = kt * 32;

        // ---- prefetch next tile's K (contiguous 4KB) and V rows ----
        if (kt + 1 < nkt) {
            const int kn = kb + 32;
            const __bf16* kp = Kbb + (size_t)kn * 64;   // 32 rows * 128B
            __builtin_prefetch(kp + lane * 64, 0, 3);
            __builtin_prefetch(kp + lane * 64 + 32, 0, 3);
#pragma unroll
            for (int dn = 0; dn < 4; dn++)
                __builtin_prefetch(Vtb + (size_t)(dn * 16 + l15) * 2048 + kn + 16 * hi, 0, 3);
        }

        // ---- S = Q · K^T (two 16-col halves, K-dim 64 = 2 WMMAs each) ----
        v8f s0 = (v8f){}, s1 = (v8f){};
        {
            const __bf16* Kr0 = Kbb + (size_t)(kb + l15) * 64;
            v16bf B00 = *(const v16bf*)(Kr0 + 16 * hi);
            v16bf B01 = *(const v16bf*)(Kr0 + 32 + 16 * hi);
            s0 = WMMA_BF16(aq0.v, B00, s0);
            s0 = WMMA_BF16(aq1.v, B01, s0);
            const __bf16* Kr1 = Kbb + (size_t)(kb + 16 + l15) * 64;
            v16bf B10 = *(const v16bf*)(Kr1 + 16 * hi);
            v16bf B11 = *(const v16bf*)(Kr1 + 32 + 16 * hi);
            s1 = WMMA_BF16(aq0.v, B10, s1);
            s1 = WMMA_BF16(aq1.v, B11, s1);
        }

        // ---- online softmax in exp2 domain (row spans 16 lanes in C-layout) ----
#pragma unroll
        for (int r = 0; r < 8; r++) {
            float a = s0[r] * SCL2;
            float c = s1[r] * SCL2;
            const int qg = qbase + r + 8 * hi;
            if (kb + l15 > qg)      a = -__builtin_inff();
            if (kb + 16 + l15 > qg) c = -__builtin_inff();
            float tmax = fmaxf(a, c);
            tmax = fmaxf(tmax, __shfl_xor(tmax, 1));
            tmax = fmaxf(tmax, __shfl_xor(tmax, 2));
            tmax = fmaxf(tmax, __shfl_xor(tmax, 4));
            tmax = fmaxf(tmax, __shfl_xor(tmax, 8));
            float mnew = fmaxf(mrow[r], tmax);
            float pa = exp2f(a - mnew);              // v_exp_f32 (native exp2)
            float pb = exp2f(c - mnew);
            float corr = exp2f(mrow[r] - mnew);
            mrow[r] = mnew;
            float sum = pa + pb;
            sum += __shfl_xor(sum, 1);
            sum += __shfl_xor(sum, 2);
            sum += __shfl_xor(sum, 4);
            sum += __shfl_xor(sum, 8);
            lrow[r] = lrow[r] * corr + sum;
            O[0][r] *= corr; O[1][r] *= corr; O[2][r] *= corr; O[3][r] *= corr;
            const int m = r + 8 * hi;                // C-layout row
            myP[m * 32 + l15]      = f2bf(pa);       // P row-major [16][32] bf16
            myP[m * 32 + 16 + l15] = f2bf(pb);
        }
        asm volatile("" ::: "memory");               // keep LDS reload after stores

        // ---- reload P in A-layout; O += P · V ----
        V16 ap;
        const __bf16* pp = (const __bf16*)myP;
        ap.h[0] = *(const v8bf*)(pp + (size_t)l15 * 32 + 16 * hi);
        ap.h[1] = *(const v8bf*)(pp + (size_t)l15 * 32 + 8 + 16 * hi);
#pragma unroll
        for (int dn = 0; dn < 4; dn++) {
            const __bf16* vp = Vtb + (size_t)(dn * 16 + l15) * 2048 + kb + 16 * hi;
            v16bf Bv = *(const v16bf*)vp;
            O[dn] = WMMA_BF16(ap.v, Bv, O[dn]);
        }
    }

    // ---- epilogue: out = O / l (write-once -> non-temporal stores) ----
    float invl[8];
#pragma unroll
    for (int r = 0; r < 8; r++) invl[r] = 1.0f / lrow[r];
#pragma unroll
    for (int dn = 0; dn < 4; dn++) {
#pragma unroll
        for (int r = 0; r < 8; r++) {
            size_t o = ((size_t)b * 2048 + qbase + r + 8 * hi) * 64 + dn * 16 + l15;
            __builtin_nontemporal_store(O[dn][r] * invl[r], &out[o]);
        }
    }
}

// ---------------------------------------------------------------------------
extern "C" void kernel_launch(void* const* d_in, const int* in_sizes, int n_in,
                              void* d_out, int out_size, void* d_ws, size_t ws_size,
                              hipStream_t stream) {
    const float* x    = (const float*)d_in[0];
    const float* Wq   = (const float*)d_in[1];
    const float* Wk   = (const float*)d_in[2];
    const float* Wv   = (const float*)d_in[3];
    const float* fcos = (const float*)d_in[4];
    const float* fsin = (const float*)d_in[5];
    float* out = (float*)d_out;

    char* ws = (char*)d_ws;
    __bf16* Wt = (__bf16*)ws;                               // 393216 B
    __bf16* Qb = (__bf16*)(ws + 393216);                    // 2 MB
    __bf16* Kb = (__bf16*)(ws + 393216 + 2097152);          // 2 MB
    __bf16* Vt = (__bf16*)(ws + 393216 + 2 * 2097152);      // 2 MB

    hipLaunchKernelGGL(wt_kernel,        dim3(768), dim3(256), 0, stream, Wq, Wk, Wv, Wt);
    hipLaunchKernelGGL(proj_rope_kernel, dim3(256), dim3(128), 0, stream,
                       x, fcos, fsin, Wt, Qb, Kb, Vt);
    hipLaunchKernelGGL(flash_attn_kernel, dim3(256), dim3(128), 0, stream,
                       Qb, Kb, Vt, out);
}
